// GRRN_GRU_15839839387964
// MI455X (gfx1250) — matrix operand order repaired and checked by reading, not compiled, not run on hardware
//
#include <hip/hip_runtime.h>
#include <math.h>
#include <stdint.h>

// ---------------------------------------------------------------------------
// Types for CDNA5 WMMA (wave32): v_wmma_f32_16x16x32_bf16
// ---------------------------------------------------------------------------
typedef __bf16 bf16;
typedef __bf16 bf16x8  __attribute__((ext_vector_type(8)));
typedef __bf16 v16bf   __attribute__((ext_vector_type(16)));
typedef float  v8f     __attribute__((ext_vector_type(8)));

static __device__ __forceinline__ bf16 f2bf(float x) {
    unsigned u = __builtin_bit_cast(unsigned, x);
    u += 0x7FFFu + ((u >> 16) & 1u);           // round-to-nearest-even
    unsigned short h = (unsigned short)(u >> 16);
    return __builtin_bit_cast(bf16, h);
}
static __device__ __forceinline__ float sigm(float x) { return 1.0f / (1.0f + expf(-x)); }

// Problem constants
#define NN 12
#define EE 7
#define SS 1024
#define TSP1 4          // TS+1 outer iterations
#define D_IN 26400
#define D_INP 26432     // padded to a multiple of 64 for the K loop
#define BB 2
#define TT 4
#define BT 8            // B*T
#define M1 96           // BT*N
#define BNN 288         // B*N*N
#define H0_STD 0.00259971f   // sqrt(2/(BNN*S + S))

// ---------------------------------------------------------------------------
// Async global -> LDS copy (gfx1250): GLOBAL_LOAD_ASYNC_TO_LDS_B128, ASYNCcnt
// LDS address = low 32 bits of flat pointer.
// ---------------------------------------------------------------------------
static __device__ __forceinline__ void async_b128(bf16* lds, const bf16* g) {
    uint32_t loff = (uint32_t)(uintptr_t)(void*)lds;
    asm volatile("global_load_async_to_lds_b128 %0, %1, off"
                 :: "v"(loff), "v"((uint64_t)(uintptr_t)g)
                 : "memory");
}
#if __has_builtin(__builtin_amdgcn_s_wait_asynccnt)
#define WAIT_ASYNC(n) __builtin_amdgcn_s_wait_asynccnt(n)
#else
#define WAIT_ASYNC(n) asm volatile("s_wait_asynccnt %0" :: "i"(n) : "memory")
#endif

// ---------------------------------------------------------------------------
// f32 -> bf16 conversion (grid-stride), plus K-padded variant
// ---------------------------------------------------------------------------
__global__ void k_f2bf(const float* __restrict__ s, bf16* __restrict__ d, int n) {
    int i = blockIdx.x * blockDim.x + threadIdx.x;
    int stride = gridDim.x * blockDim.x;
    for (; i < n; i += stride) d[i] = f2bf(s[i]);
}
__global__ void k_f2bf_pad(const float* __restrict__ s, bf16* __restrict__ d,
                           int rows, int K, int Kp) {
    int i = blockIdx.x * blockDim.x + threadIdx.x;
    int total = rows * Kp;
    int stride = gridDim.x * blockDim.x;
    for (; i < total; i += stride) {
        int r = i / Kp, c = i % Kp;
        d[i] = (c < K) ? f2bf(s[(size_t)r * K + c]) : f2bf(0.0f);
    }
}

// ---------------------------------------------------------------------------
// Generic bf16 WMMA GEMM:  C[M,Nc] = act( A[M,K] @ W[Nc,K]^T + bias )
// A row-major (lda), W row-major (ldw), K % 64 == 0.
// Block: 256 threads = 8 waves (wave32). Workgroup tile 64(M) x 128(N),
// wave tile 32x32 = 2x2 WMMA 16x16x32 tiles, K step 64 (2 chunks).
// Global->LDS via async-to-LDS b128 with double buffering (ASYNCcnt).
// Row indices are clamped (not branched); OOB results masked at store.
// ---------------------------------------------------------------------------
#define BM 64
#define BN 128
#define BK 64
#define LDSA 72   // 64 + 8 bf16 pad (16B-aligned rows)
#define LDSB 72

static __device__ __forceinline__ void issue_tile_async(
    const bf16* __restrict__ A, int lda, const bf16* __restrict__ W, int ldw,
    int M, int Nc, int bm0, int bn0, int k0, int tid,
    bf16* __restrict__ As, bf16* __restrict__ Ws) {
    // A tile: 64 rows x 64 cols, 16 bf16/thread => 2 async b128
    {
        const int r = tid >> 2, q = tid & 3;
        int gr = bm0 + r; if (gr >= M) gr = M - 1;          // clamp, no branch
        const bf16* g = A + (size_t)gr * lda + k0 + q * 16;
        bf16* l = As + r * LDSA + q * 16;
        async_b128(l,     g);
        async_b128(l + 8, g + 8);
        __builtin_prefetch(g + BK, 0, 1);                    // next-next tile hint
    }
    // W tile: 128 rows x 64 cols, 32 bf16/thread => 4 async b128
    {
        const int r = tid >> 1, h = tid & 1;
        int gr = bn0 + r; if (gr >= Nc) gr = Nc - 1;
        const bf16* g = W + (size_t)gr * ldw + k0 + h * 32;
        bf16* l = Ws + r * LDSB + h * 32;
        async_b128(l,      g);
        async_b128(l + 8,  g + 8);
        async_b128(l + 16, g + 16);
        async_b128(l + 24, g + 24);
        __builtin_prefetch(g + BK, 0, 1);
    }
}

__global__ __launch_bounds__(256)
void k_gemm(const bf16* __restrict__ A, int lda,
            const bf16* __restrict__ W, int ldw,
            const float* __restrict__ bias,
            float* __restrict__ C, int ldc,
            bf16* __restrict__ Cb,
            int M, int Nc, int K, int act) {
    __shared__ __align__(16) bf16 AsBuf[2][BM * LDSA];
    __shared__ __align__(16) bf16 WsBuf[2][BN * LDSB];

    const int tid  = threadIdx.x;
    const int lane = tid & 31;
    const int wave = tid >> 5;
    const int wm = wave >> 2;          // 0..1
    const int wn = wave & 3;           // 0..3
    const int bm0 = blockIdx.y * BM;
    const int bn0 = blockIdx.x * BN;
    const int lh = lane >> 4;          // half-wave select
    const int ll = lane & 15;

    v8f acc[2][2] = {};

    const int nt = K >> 6;             // tiles of 64
    issue_tile_async(A, lda, W, ldw, M, Nc, bm0, bn0, 0, tid, AsBuf[0], WsBuf[0]);

    for (int t = 0; t < nt; ++t) {
        const bf16* Ab = AsBuf[t & 1];
        const bf16* Wb = WsBuf[t & 1];
        if (t + 1 < nt) {
            issue_tile_async(A, lda, W, ldw, M, Nc, bm0, bn0, (t + 1) << 6, tid,
                             AsBuf[(t + 1) & 1], WsBuf[(t + 1) & 1]);
            WAIT_ASYNC(6);             // only the 6 just-issued may remain in flight
        } else {
            WAIT_ASYNC(0);
        }
        __syncthreads();               // tile t resident for all waves

        // --- two K=32 chunks, 8 WMMA per tile ---
#pragma unroll
        for (int kk = 0; kk < 2; ++kk) {
            const int kc = kk * 32;
            v16bf afrag[2], bfrag[2];
#pragma unroll
            for (int mi = 0; mi < 2; ++mi) {
                const int row = wm * 32 + mi * 16 + ll;
                const int kb  = kc + lh * 8;
                bf16x8 lo = *(const bf16x8*)&Ab[row * LDSA + kb];
                bf16x8 hi = *(const bf16x8*)&Ab[row * LDSA + kb + 16];
#pragma unroll
                for (int i = 0; i < 8; ++i) { afrag[mi][i] = lo[i]; afrag[mi][8 + i] = hi[i]; }
            }
#pragma unroll
            for (int nj = 0; nj < 2; ++nj) {
                const int row = wn * 32 + nj * 16 + ll;
                const int kb  = kc + lh * 8;
                bf16x8 lo = *(const bf16x8*)&Wb[row * LDSB + kb];
                bf16x8 hi = *(const bf16x8*)&Wb[row * LDSB + kb + 16];
#pragma unroll
                for (int i = 0; i < 8; ++i) { bfrag[nj][i] = lo[i]; bfrag[nj][8 + i] = hi[i]; }
            }
#pragma unroll
            for (int mi = 0; mi < 2; ++mi)
#pragma unroll
                for (int nj = 0; nj < 2; ++nj)
                    acc[mi][nj] = __builtin_amdgcn_wmma_f32_16x16x32_bf16(
                        false, afrag[mi], false, bfrag[nj],
                        (short)0, acc[mi][nj], false, false);
        }
        __syncthreads();               // all waves done reading before buffer reuse
    }

    // --- epilogue: C/D layout (VGPR r: M=r / M=8+r per half-wave, N=lane&15)
#pragma unroll
    for (int mi = 0; mi < 2; ++mi) {
#pragma unroll
        for (int nj = 0; nj < 2; ++nj) {
#pragma unroll
            for (int r = 0; r < 8; ++r) {
                const int row = bm0 + wm * 32 + mi * 16 + lh * 8 + r;
                const int col = bn0 + wn * 32 + nj * 16 + ll;
                if (row < M && col < Nc) {
                    float v = acc[mi][nj][r];
                    if (bias) v += bias[col];
                    if (act == 1) v = sigm(v);
                    else if (act == 2) v = tanhf(v);
                    C[(size_t)row * ldc + col] = v;
                    if (Cb) Cb[(size_t)row * ldc + col] = f2bf(v);
                }
            }
        }
    }
}

// ---------------------------------------------------------------------------
// cvec[j] = bcomp @ Wih[j,:] + bih[j]   (3072 outputs, K=1024)
// ---------------------------------------------------------------------------
__global__ void k_cvec(const float* __restrict__ bcomp, const float* __restrict__ Wih,
                       const float* __restrict__ bih, float* __restrict__ cvec) {
    int j = blockIdx.x * blockDim.x + threadIdx.x;
    if (j >= 3 * SS) return;
    float s = bih[j];
    const float* wr = Wih + (size_t)j * SS;
    for (int k = 0; k < SS; ++k) s += bcomp[k] * wr[k];
    cvec[j] = s;
}

// ---------------------------------------------------------------------------
// h0 init: deterministic hash-based pseudo-normal * H0_STD  (288*1024 elems)
// ---------------------------------------------------------------------------
__global__ void k_h0(float* __restrict__ h, bf16* __restrict__ hb, int seed) {
    int i = blockIdx.x * blockDim.x + threadIdx.x;
    if (i >= BNN * SS) return;
    uint32_t x = (uint32_t)i * 2654435761u ^ ((uint32_t)seed * 0x9E3779B9u + 0x85EBCA6Bu);
    x ^= x >> 16; x *= 0x7FEB352Du; x ^= x >> 15; x *= 0x846CA68Bu; x ^= x >> 16;
    float u = (float)(x >> 8) * (1.0f / 16777216.0f);      // [0,1)
    float val = (u * 2.0f - 1.0f) * 1.7320508f * H0_STD;   // unit-variance uniform
    h[i] = val;
    hb[i] = f2bf(val);
}

// ---------------------------------------------------------------------------
// Fused GRU gates. Row = (b, n, m) in [0,288); col in [0,1024).
// gi composed on-the-fly: gi = P[row_n] + Q[row_m] + cvec ; gh includes bhh.
// h <- (1-z)*tanh(gi_n + r*gh_n) + z*h
// ---------------------------------------------------------------------------
__global__ void k_gru_gate(const float* __restrict__ P, const float* __restrict__ Q,
                           const float* __restrict__ cvec, const float* __restrict__ gh,
                           float* __restrict__ h, bf16* __restrict__ hb, int tt) {
    const int col = blockIdx.x * blockDim.x + threadIdx.x;   // grid.x = 4
    const int row = blockIdx.y;                              // 288
    const int b = row / (NN * NN);
    const int nm = row % (NN * NN);
    const int n = nm / NN, m = nm % NN;
    const size_t pn = (size_t)((b * TT + tt) * NN + n) * (3 * SS);
    const size_t qm = (size_t)((b * TT + tt) * NN + m) * (3 * SS);
    const size_t gr = (size_t)row * (3 * SS);

    float xr = P[pn + col]          + Q[qm + col]          + cvec[col]          + gh[gr + col];
    float xz = P[pn + SS + col]     + Q[qm + SS + col]     + cvec[SS + col]     + gh[gr + SS + col];
    float xn = P[pn + 2 * SS + col] + Q[qm + 2 * SS + col] + cvec[2 * SS + col];
    float hn = gh[gr + 2 * SS + col];

    float r = sigm(xr);
    float z = sigm(xz);
    float nn2 = tanhf(xn + r * hn);
    float hv = h[(size_t)row * SS + col];
    float hnew = (1.0f - z) * nn2 + z * hv;
    h[(size_t)row * SS + col] = hnew;
    hb[(size_t)row * SS + col] = f2bf(hnew);
}

// ---------------------------------------------------------------------------
// Per-row score: gs[bt,n,m,e] = sigmoid( dot(h_row, Wa[e]) + ba[e] )
// One wave32 per row; __shfl_down reduction.
// ---------------------------------------------------------------------------
__global__ __launch_bounds__(256)
void k_rowscore(const float* __restrict__ h, const float* __restrict__ Wa,
                const float* __restrict__ ba, float* __restrict__ gs, int tt, int e) {
    const int lane = threadIdx.x & 31;
    const int wave = threadIdx.x >> 5;
    const int row = blockIdx.x * 8 + wave;
    if (row >= BNN) return;
    const float* wv = Wa + (size_t)e * SS;
    float s = 0.0f;
    for (int c = lane; c < SS; c += 32) s += h[(size_t)row * SS + c] * wv[c];
#pragma unroll
    for (int off = 16; off > 0; off >>= 1) s += __shfl_down(s, off, 32);
    if (lane == 0) {
        const int b = row / (NN * NN);
        const int nm = row % (NN * NN);
        const int n = nm / NN, m = nm % NN;
        const int bt = b * TT + tt;
        gs[((size_t)(bt * NN + n) * NN + m) * EE + e] = sigm(s + ba[e]);
    }
}

// ---------------------------------------------------------------------------
// merged[bt,n,s] = sum_{e,m} gs[bt,n,m,e] * msgs_e[bt*12+m, s]
// Also builds a_bf = concat(merged, prop) in bf16.
// ---------------------------------------------------------------------------
__global__ void k_merged(const float* __restrict__ gs, const float* __restrict__ msgs,
                         const float* __restrict__ prop, float* __restrict__ merged,
                         bf16* __restrict__ a_bf) {
    const int s = blockIdx.x * blockDim.x + threadIdx.x;   // grid.x = 4
    const int row = blockIdx.y;                            // 96 = bt*12+n
    const int bt = row / NN, n = row % NN;
    float acc = 0.0f;
    for (int e = 0; e < EE; ++e) {
        const float* me = msgs + (size_t)e * M1 * SS;
        for (int m = 0; m < NN; ++m) {
            float g = gs[((size_t)(bt * NN + n) * NN + m) * EE + e];
            acc += g * me[(size_t)(bt * NN + m) * SS + s];
        }
    }
    merged[(size_t)row * SS + s] = acc;
    a_bf[(size_t)row * 2 * SS + s] = f2bf(acc);
    a_bf[(size_t)row * 2 * SS + SS + s] = f2bf(prop[(size_t)row * SS + s]);
}

// a2 = concat(merged, r*prop) in bf16
__global__ void k_a2(const float* __restrict__ merged, const float* __restrict__ rg,
                     const float* __restrict__ prop, bf16* __restrict__ a2) {
    const int s = blockIdx.x * blockDim.x + threadIdx.x;
    const int row = blockIdx.y;
    a2[(size_t)row * 2 * SS + s] = f2bf(merged[(size_t)row * SS + s]);
    a2[(size_t)row * 2 * SS + SS + s] =
        f2bf(rg[(size_t)row * SS + s] * prop[(size_t)row * SS + s]);
}

// prop <- (1-z)*prop + z*h_hat  (f32 + bf16 mirror)
__global__ void k_prop_update(const float* __restrict__ z, const float* __restrict__ hh,
                              float* __restrict__ prop, bf16* __restrict__ propb) {
    const int i = blockIdx.x * blockDim.x + threadIdx.x;
    if (i >= M1 * SS) return;
    float zz = z[i];
    float p = (1.0f - zz) * prop[i] + zz * hh[i];
    prop[i] = p;
    propb[i] = f2bf(p);
}

// ---------------------------------------------------------------------------
// Final scoring: action_score [672] + acty_score [48] + actor_score [1152]
// into d_out concatenated in return order.  Single block.
// ---------------------------------------------------------------------------
__global__ __launch_bounds__(256)
void k_score(const float* __restrict__ gs, float* __restrict__ out) {
    __shared__ float action[BT * NN * EE];  // 672
    __shared__ float asum[BT * NN];         // 96
    __shared__ float tmp[BT * EE];          // 56
    const int tid = threadIdx.x;

    for (int idx = tid; idx < BT * NN * EE; idx += 256) {
        const int e = idx % EE;
        const int n = (idx / EE) % NN;
        const int bt = idx / (EE * NN);
        float s = 0.0f;
        for (int m = 0; m < NN; ++m) {
            float g0 = gs[((size_t)(bt * NN + n) * NN + m) * EE + 0];
            float ge = gs[((size_t)(bt * NN + n) * NN + m) * EE + e];
            s += (1.0f - g0) * ge;
        }
        action[idx] = s;
        out[idx] = s;
    }
    for (int idx = tid; idx < BT * NN * NN; idx += 256) {
        const int m = idx % NN;
        const int n = (idx / NN) % NN;
        const int bt = idx / (NN * NN);
        out[720 + idx] = 1.0f - gs[((size_t)(bt * NN + n) * NN + m) * EE + 0];
    }
    for (int idx = tid; idx < BT * NN; idx += 256) {
        const int n = idx % NN, bt = idx / NN;
        float s = 0.0f;
        for (int m = 0; m < NN; ++m)
            s += 1.0f - gs[((size_t)(bt * NN + n) * NN + m) * EE + 0];
        asum[idx] = s;
    }
    __syncthreads();
    for (int idx = tid; idx < BT * EE; idx += 256) {
        const int e = idx % EE, bt = idx / EE;
        float s = 0.0f;
        for (int n = 0; n < NN; ++n)
            s += action[(bt * NN + n) * EE + e] * asum[bt * NN + n];
        tmp[idx] = s;
    }
    __syncthreads();
    for (int bt = tid; bt < BT; bt += 256) {
        float mx = -1e30f;
        for (int e = 1; e < EE; ++e) mx = fmaxf(mx, tmp[bt * EE + e]);
        float ex[EE - 1];
        float den = 0.0f;
        for (int e = 1; e < EE; ++e) { ex[e - 1] = expf(tmp[bt * EE + e] - mx); den += ex[e - 1]; }
        for (int e = 1; e < EE; ++e) out[672 + bt * (EE - 1) + (e - 1)] = ex[e - 1] / den;
    }
}

// ---------------------------------------------------------------------------
// Host orchestration
// ---------------------------------------------------------------------------
static inline void* bump(char*& p, size_t bytes) {
    void* r = (void*)p;
    p += (bytes + 255) & ~(size_t)255;
    return r;
}

static inline void launch_gemm(const bf16* A, int lda, const bf16* W, int ldw,
                               const float* bias, float* C, int ldc, bf16* Cb,
                               int M, int Nc, int K, int act, hipStream_t s) {
    dim3 grid((Nc + BN - 1) / BN, (M + BM - 1) / BM);
    k_gemm<<<grid, dim3(256), 0, s>>>(A, lda, W, ldw, bias, C, ldc, Cb, M, Nc, K, act);
}

static inline void launch_f2bf(const float* src, bf16* dst, int n, hipStream_t s) {
    int blocks = (n + 255) / 256;
    if (blocks > 65535) blocks = 65535;
    k_f2bf<<<blocks, 256, 0, s>>>(src, dst, n);
}

extern "C" void kernel_launch(void* const* d_in, const int* in_sizes, int n_in,
                              void* d_out, int out_size, void* d_ws, size_t ws_size,
                              hipStream_t stream) {
    (void)in_sizes; (void)n_in; (void)out_size; (void)ws_size;
    const float* pose  = (const float*)d_in[0];
    const float* Wc    = (const float*)d_in[5];
    const float* bc    = (const float*)d_in[6];
    const float* We    = (const float*)d_in[7];
    const float* be    = (const float*)d_in[8];
    const float* Wcomp = (const float*)d_in[9];
    const float* bcomp = (const float*)d_in[10];
    const float* Wr    = (const float*)d_in[11];
    const float* br    = (const float*)d_in[12];
    const float* Wz    = (const float*)d_in[13];
    const float* bz    = (const float*)d_in[14];
    const float* Wh    = (const float*)d_in[15];
    const float* bh    = (const float*)d_in[16];
    const float* Wih   = (const float*)d_in[17];
    const float* Whh   = (const float*)d_in[18];
    const float* bih   = (const float*)d_in[19];
    const float* bhh   = (const float*)d_in[20];
    const float* Wa    = (const float*)d_in[21];
    const float* ba    = (const float*)d_in[22];
    float* out = (float*)d_out;

    // ---- workspace layout (bump allocator, 256B aligned) ----
    char* p = (char*)d_ws;
    bf16* WcB    = (bf16*)bump(p, sizeof(bf16) * (size_t)SS * D_INP);
    bf16* poseB  = (bf16*)bump(p, sizeof(bf16) * (size_t)M1 * D_INP);
    bf16* WeB    = (bf16*)bump(p, sizeof(bf16) * (size_t)EE * SS * SS);
    bf16* WcompB = (bf16*)bump(p, sizeof(bf16) * (size_t)SS * 2 * SS);
    bf16* WrB    = (bf16*)bump(p, sizeof(bf16) * (size_t)SS * 2 * SS);
    bf16* WzB    = (bf16*)bump(p, sizeof(bf16) * (size_t)SS * 2 * SS);
    bf16* WhB    = (bf16*)bump(p, sizeof(bf16) * (size_t)SS * 2 * SS);
    bf16* WihB   = (bf16*)bump(p, sizeof(bf16) * (size_t)3 * SS * SS);
    bf16* WhhB   = (bf16*)bump(p, sizeof(bf16) * (size_t)3 * SS * SS);
    float* propF = (float*)bump(p, sizeof(float) * (size_t)M1 * SS);
    bf16*  propB = (bf16*) bump(p, sizeof(bf16)  * (size_t)M1 * SS);
    float* msgsF = (float*)bump(p, sizeof(float) * (size_t)EE * M1 * SS);
    bf16*  msgsB = (bf16*) bump(p, sizeof(bf16)  * (size_t)EE * M1 * SS);
    float* uF    = (float*)bump(p, sizeof(float) * (size_t)M1 * SS);
    float* vF    = (float*)bump(p, sizeof(float) * (size_t)M1 * SS);
    bf16*  uB    = (bf16*) bump(p, sizeof(bf16)  * (size_t)M1 * SS);
    bf16*  vB    = (bf16*) bump(p, sizeof(bf16)  * (size_t)M1 * SS);
    float* Pm    = (float*)bump(p, sizeof(float) * (size_t)M1 * 3 * SS);
    float* Qm    = (float*)bump(p, sizeof(float) * (size_t)M1 * 3 * SS);
    float* cvec  = (float*)bump(p, sizeof(float) * (size_t)3 * SS);
    float* hF    = (float*)bump(p, sizeof(float) * (size_t)BNN * SS);
    bf16*  hB    = (bf16*) bump(p, sizeof(bf16)  * (size_t)BNN * SS);
    float* ghF   = (float*)bump(p, sizeof(float) * (size_t)BNN * 3 * SS);
    float* gsF   = (float*)bump(p, sizeof(float) * (size_t)BT * NN * NN * EE);
    float* mergF = (float*)bump(p, sizeof(float) * (size_t)M1 * SS);
    bf16*  aB    = (bf16*) bump(p, sizeof(bf16)  * (size_t)M1 * 2 * SS);
    bf16*  a2B   = (bf16*) bump(p, sizeof(bf16)  * (size_t)M1 * 2 * SS);
    float* rF    = (float*)bump(p, sizeof(float) * (size_t)M1 * SS);
    float* zF    = (float*)bump(p, sizeof(float) * (size_t)M1 * SS);
    float* hhF   = (float*)bump(p, sizeof(float) * (size_t)M1 * SS);

    // ---- weight + input conversions to bf16 (K-padded where K%64 != 0) ----
    {
        int blocks = 65535;
        k_f2bf_pad<<<blocks, 256, 0, stream>>>(Wc,   WcB,   SS, D_IN, D_INP);
        k_f2bf_pad<<<(M1 * D_INP + 255) / 256, 256, 0, stream>>>(pose, poseB, M1, D_IN, D_INP);
    }
    launch_f2bf(We,    WeB,    EE * SS * SS,   stream);
    launch_f2bf(Wcomp, WcompB, SS * 2 * SS,    stream);
    launch_f2bf(Wr,    WrB,    SS * 2 * SS,    stream);
    launch_f2bf(Wz,    WzB,    SS * 2 * SS,    stream);
    launch_f2bf(Wh,    WhB,    SS * 2 * SS,    stream);
    launch_f2bf(Wih,   WihB,   3 * SS * SS,    stream);
    launch_f2bf(Whh,   WhhB,   3 * SS * SS,    stream);

    // cvec = bcomp @ Wih.T + bih
    k_cvec<<<(3 * SS + 255) / 256, 256, 0, stream>>>(bcomp, Wih, bih, cvec);

    // prop = pose @ Wc.T + bc
    launch_gemm(poseB, D_INP, WcB, D_INP, bc, propF, SS, propB, M1, SS, D_INP, 0, stream);

    // ---- outer refinement loop (TS+1 = 4 iterations) ----
    for (int ot = 0; ot < TSP1; ++ot) {
        for (int e = 0; e < EE; ++e) {
            const size_t eoff = (size_t)e * M1 * SS;
            // msgs_e = prop @ We[e].T + be[e]
            launch_gemm(propB, SS, WeB + (size_t)e * SS * SS, SS, be + e * SS,
                        msgsF + eoff, SS, msgsB + eoff, M1, SS, SS, 0, stream);
            // u = rf @ Wcomp[:, :S].T ; v = rf @ Wcomp[:, S:].T
            launch_gemm(msgsB + eoff, SS, WcompB,      2 * SS, nullptr, uF, SS, uB, M1, SS, SS, 0, stream);
            launch_gemm(msgsB + eoff, SS, WcompB + SS, 2 * SS, nullptr, vF, SS, vB, M1, SS, SS, 0, stream);
            // P = u @ Wih.T ; Q = v @ Wih.T
            launch_gemm(uB, SS, WihB, SS, nullptr, Pm, 3 * SS, nullptr, M1, 3 * SS, SS, 0, stream);
            launch_gemm(vB, SS, WihB, SS, nullptr, Qm, 3 * SS, nullptr, M1, 3 * SS, SS, 0, stream);
            // h0 for this (ot,e)
            k_h0<<<(BNN * SS + 255) / 256, 256, 0, stream>>>(hF, hB, ot * EE + e);
            // GRU scan over T=4 steps
            for (int tt = 0; tt < TT; ++tt) {
                launch_gemm(hB, SS, WhhB, SS, bhh, ghF, 3 * SS, nullptr, BNN, 3 * SS, SS, 0, stream);
                k_gru_gate<<<dim3(SS / 256, BNN), 256, 0, stream>>>(Pm, Qm, cvec, ghF, hF, hB, tt);
                k_rowscore<<<(BNN + 7) / 8, 256, 0, stream>>>(hF, Wa, ba, gsF, tt, e);
            }
        }
        // merged + a = concat(merged, prop)
        k_merged<<<dim3(SS / 256, M1), 256, 0, stream>>>(gsF, msgsF, propF, mergF, aB);
        // r, z gates
        launch_gemm(aB, 2 * SS, WrB, 2 * SS, br, rF, SS, nullptr, M1, SS, 2 * SS, 1, stream);
        launch_gemm(aB, 2 * SS, WzB, 2 * SS, bz, zF, SS, nullptr, M1, SS, 2 * SS, 1, stream);
        // a2 = concat(merged, r*prop); h_hat = tanh(a2 @ Wh.T + bh)
        k_a2<<<dim3(SS / 256, M1), 256, 0, stream>>>(mergF, rF, propF, a2B);
        launch_gemm(a2B, 2 * SS, WhB, 2 * SS, bh, hhF, SS, nullptr, M1, SS, 2 * SS, 2, stream);
        // prop = (1-z)*prop + z*h_hat
        k_prop_update<<<(M1 * SS + 255) / 256, 256, 0, stream>>>(zF, hhF, propF, propB);
    }

    // ---- final scores ----
    k_score<<<1, 256, 0, stream>>>(gsF, out);
}
